// EGNN_72456098283795
// MI455X (gfx1250) — matrix operand order repaired and checked by reading, not compiled, not run on hardware
//
#include <hip/hip_runtime.h>
#include <hip/hip_bf16.h>

typedef __attribute__((ext_vector_type(16))) _Float16 v16h;
typedef __attribute__((ext_vector_type(8)))  _Float16 v8h;
typedef __attribute__((ext_vector_type(8)))  float    v8f;

#define HID 128
#define PW 136   // padded LDS row (halves): 272B = 17*16B -> 16B aligned, +4 banks/row

// ---------------------------------------------------------------- utilities
__global__ void k_zero(float* __restrict__ p, long n) {
  long i = (long)blockIdx.x * blockDim.x + threadIdx.x;
  long s = (long)gridDim.x * blockDim.x;
  for (; i < n; i += s) p[i] = 0.f;
}

__global__ void k_countf(const int* __restrict__ ids, long n, float* __restrict__ cnt) {
  long i = (long)blockIdx.x * blockDim.x + threadIdx.x;
  if (i < n) atomicAdd(&cnt[ids[i]], 1.0f);
}

__global__ void k_nrm(float* __restrict__ deg, long n) {
  long i = (long)blockIdx.x * blockDim.x + threadIdx.x;
  if (i < n) deg[i] = rsqrtf(deg[i] + 1.0f);
}

// per-edge coefficient nrm[src]*nrm[dst] (structure constant)
__global__ void k_ecoef(const int* __restrict__ src, const int* __restrict__ dst, long E,
                        const float* __restrict__ nrm, float* __restrict__ coef) {
  long e = (long)blockIdx.x * blockDim.x + threadIdx.x;
  if (e < E) coef[e] = nrm[src[e]] * nrm[dst[e]];
}

// ---------------------------------------------------------------- encoder
__global__ __launch_bounds__(128)
void k_encode(const float* __restrict__ X, int K,
              const float* __restrict__ W, const float* __restrict__ b,
              float* __restrict__ H) {
  __shared__ float xs[96];
  const int node = blockIdx.x;
  const int c = threadIdx.x;
  for (int k = c; k < K; k += 128) xs[k] = X[(long)node * K + k];
  __syncthreads();
  float acc = b[c];
  for (int k = 0; k < K; ++k) acc = fmaf(xs[k], W[k * HID + c], acc);
  H[(long)node * HID + c] = acc;
}

// ---------------------------------------------------------------- WMMA GEMM
// C[M,128] (+)= (A[M,128] (+ vx[batch[row]])) @ W[128,128] (+ bias)
// block = 256 threads = 8 waves; block tile = 64 rows x 128 cols.
// wave w owns output columns [16w,16w+16); per K-step the B fragment is
// loaded once (2x ds_load_b128 from transposed, padded W) and reused over
// 4 M-tiles -> 16 v_wmma_f32_16x16x32_f16 per wave.
__global__ __launch_bounds__(256)
void k_gemm128(const float* __restrict__ A,
               const float* __restrict__ W,
               const float* __restrict__ bias,   // nullable
               const float* __restrict__ vx,     // nullable [B][128]
               const int*   __restrict__ batch,  // used iff vx
               float* __restrict__ C,
               int accumulate) {
  __shared__ _Float16 Ah[64][PW];      // 17.0 KB
  __shared__ _Float16 Wt[HID][PW];     // 34.0 KB (transposed: Wt[n][k])
  const int t = threadIdx.x;
  const long row0 = (long)blockIdx.x * 64;

  // stage W transposed: Wt[n][k] = W[k*128+n]
  const float4* W4 = (const float4*)W;
  for (int i = t; i < HID * 32; i += 256) {
    int k = i >> 5, c4 = (i & 31) * 4;
    float4 wv = W4[i];
    Wt[c4 + 0][k] = (_Float16)wv.x;
    Wt[c4 + 1][k] = (_Float16)wv.y;
    Wt[c4 + 2][k] = (_Float16)wv.z;
    Wt[c4 + 3][k] = (_Float16)wv.w;
  }
  // stage A tile (+ virtual-node broadcast)
  for (int i = t; i < 64 * 32; i += 256) {
    int r = i >> 5, c4 = (i & 31) * 4;
    float4 av = ((const float4*)(A + (row0 + r) * HID))[i & 31];
    if (vx) {
      float4 vv = ((const float4*)(vx + (long)batch[row0 + r] * HID))[i & 31];
      av.x += vv.x; av.y += vv.y; av.z += vv.z; av.w += vv.w;
    }
    Ah[r][c4 + 0] = (_Float16)av.x;
    Ah[r][c4 + 1] = (_Float16)av.y;
    Ah[r][c4 + 2] = (_Float16)av.z;
    Ah[r][c4 + 3] = (_Float16)av.w;
  }
  __syncthreads();

  const int lane = t & 31, wave = t >> 5;
  const int m  = lane & 15;                 // row within 16-row M-tile
  const int kh = lane >> 4;                 // K-half selector
  const int n  = wave * 16 + (lane & 15);   // output column

  v8f acc[4] = {{}, {}, {}, {}};
#pragma unroll
  for (int kt = 0; kt < 4; ++kt) {
    const int k0 = kt * 32;
    v16h bm;
    {
      v8h b0 = *(const v8h*)&Wt[n][k0 + kh * 8];
      v8h b1 = *(const v8h*)&Wt[n][k0 + 16 + kh * 8];
#pragma unroll
      for (int j = 0; j < 8; ++j) { bm[j] = b0[j]; bm[j + 8] = b1[j]; }
    }
#pragma unroll
    for (int mt = 0; mt < 4; ++mt) {
      v16h a;
      v8h a0 = *(const v8h*)&Ah[mt * 16 + m][k0 + kh * 8];
      v8h a1 = *(const v8h*)&Ah[mt * 16 + m][k0 + 16 + kh * 8];
#pragma unroll
      for (int j = 0; j < 8; ++j) { a[j] = a0[j]; a[j + 8] = a1[j]; }
      acc[mt] = __builtin_amdgcn_wmma_f32_16x16x32_f16(
          false, a, false, bm, (short)0, acc[mt], false, false);
    }
  }
#pragma unroll
  for (int mt = 0; mt < 4; ++mt) {
#pragma unroll
    for (int r = 0; r < 8; ++r) {
      long idx = (row0 + mt * 16 + r + 8 * kh) * HID + n;  // C: VGPR r -> M=r+8*kh
      float v = acc[mt][r];
      if (bias) v += bias[n];
      if (accumulate) v += C[idx];
      C[idx] = v;
    }
  }
}

// ---------------------------------------------------------------- GCN pieces
__global__ void k_deg(const int* __restrict__ dst, long E, float* __restrict__ deg) {
  long e = (long)blockIdx.x * blockDim.x + threadIdx.x;
  if (e < E) atomicAdd(&deg[dst[e]], 1.0f);
}

// h = hw * nrm^2 + bias    (self loop + bias of GCNConv), float4
__global__ void k_self(const float* __restrict__ hw, const float* __restrict__ nrm,
                       const float* __restrict__ b, float* __restrict__ h, long n32) {
  long i = (long)blockIdx.x * blockDim.x + threadIdx.x;
  if (i < n32) {
    long v = i >> 5; int c4 = (int)(i & 31);
    float s = nrm[v]; s *= s;
    float4 x = ((const float4*)hw)[i];
    float4 bb = ((const float4*)b)[c4];
    float4 y;
    y.x = x.x * s + bb.x; y.y = x.y * s + bb.y;
    y.z = x.z * s + bb.z; y.w = x.w * s + bb.w;
    ((float4*)h)[i] = y;
  }
}

__global__ void k_edges(const int* __restrict__ src, const int* __restrict__ dst,
                        const float* __restrict__ coef, long E,
                        const float* __restrict__ hw, float* __restrict__ out) {
  long i = (long)blockIdx.x * blockDim.x + threadIdx.x;
  if (i < E * 32) {
    long e = i >> 5; int c4 = (int)(i & 31);
    int s = src[e], d = dst[e];
    float cf = coef[e];
    float4 v = ((const float4*)(hw + (long)s * HID))[c4];
    float* o = out + (long)d * HID + c4 * 4;
    atomicAdd(o + 0, v.x * cf);
    atomicAdd(o + 1, v.y * cf);
    atomicAdd(o + 2, v.z * cf);
    atomicAdd(o + 3, v.w * cf);
  }
}

// ---------------------------------------------------------------- batchnorm
__global__ __launch_bounds__(128)
void k_bnstats(const float* __restrict__ h, int N, float* __restrict__ stats) {
  const int c = threadIdx.x;
  const long r0 = (long)blockIdx.x * 64;
  float s = 0.f, s2 = 0.f;
  for (int i = 0; i < 64; ++i) {
    long r = r0 + i;
    if (r < N) { float v = h[r * HID + c]; s += v; s2 += v * v; }
  }
  atomicAdd(&stats[c], s);
  atomicAdd(&stats[HID + c], s2);
}

__global__ void k_bnapply(float* __restrict__ h, long n32, float invN,
                          const float* __restrict__ stats,
                          const float* __restrict__ g, const float* __restrict__ b,
                          int relu) {
  long i = (long)blockIdx.x * blockDim.x + threadIdx.x;
  if (i < n32) {
    int c4 = (int)(i & 31);
    float4 sm = ((const float4*)stats)[c4];
    float4 sq = ((const float4*)stats)[32 + c4];
    float4 gg = ((const float4*)g)[c4];
    float4 bb = ((const float4*)b)[c4];
    float4 x = ((const float4*)h)[i];
    float4 y;
    {
      float m = sm.x * invN, v = sq.x * invN - m * m;
      y.x = (x.x - m) * rsqrtf(v + 1e-5f) * gg.x + bb.x;
      m = sm.y * invN; v = sq.y * invN - m * m;
      y.y = (x.y - m) * rsqrtf(v + 1e-5f) * gg.y + bb.y;
      m = sm.z * invN; v = sq.z * invN - m * m;
      y.z = (x.z - m) * rsqrtf(v + 1e-5f) * gg.z + bb.z;
      m = sm.w * invN; v = sq.w * invN - m * m;
      y.w = (x.w - m) * rsqrtf(v + 1e-5f) * gg.w + bb.w;
    }
    if (relu) {
      y.x = fmaxf(y.x, 0.f); y.y = fmaxf(y.y, 0.f);
      y.z = fmaxf(y.z, 0.f); y.w = fmaxf(y.w, 0.f);
    }
    ((float4*)h)[i] = y;
  }
}

// ---------------------------------------------------------------- segments
// sum[si[e]][:] += h[sn[e]][:]
__global__ void k_gsadd(const int* __restrict__ sn, const int* __restrict__ si, long S,
                        const float* __restrict__ h, float* __restrict__ sum) {
  long i = (long)blockIdx.x * blockDim.x + threadIdx.x;
  if (i < S * 32) {
    long e = i >> 5; int c4 = (int)(i & 31);
    float4 v = ((const float4*)(h + (long)sn[e] * HID))[c4];
    float* o = sum + (long)si[e] * HID + c4 * 4;
    atomicAdd(o + 0, v.x); atomicAdd(o + 1, v.y);
    atomicAdd(o + 2, v.z); atomicAdd(o + 3, v.w);
  }
}

// out[seg[v]][:] += x[v][:]
__global__ void k_segsum(const int* __restrict__ seg, long n32,
                         const float* __restrict__ x, float* __restrict__ out) {
  long i = (long)blockIdx.x * blockDim.x + threadIdx.x;
  if (i < n32) {
    long v = i >> 5; int c4 = (int)(i & 31);
    float4 xv = ((const float4*)x)[i];
    float* o = out + (long)seg[v] * HID + c4 * 4;
    atomicAdd(o + 0, xv.x); atomicAdd(o + 1, xv.y);
    atomicAdd(o + 2, xv.z); atomicAdd(o + 3, xv.w);
  }
}

__global__ void k_divcnt(float* __restrict__ sum, const float* __restrict__ cnt, long n32) {
  long i = (long)blockIdx.x * blockDim.x + threadIdx.x;
  if (i < n32) {
    float r = 1.0f / fmaxf(cnt[i >> 5], 1.0f);
    float4 v = ((const float4*)sum)[i];
    v.x *= r; v.y *= r; v.z *= r; v.w *= r;
    ((float4*)sum)[i] = v;
  }
}

// ---------------------------------------------------------------- small ops
__global__ void k_vxinit(const float* __restrict__ emb, const float* __restrict__ feat,
                         float* __restrict__ vx, long n32) {
  long i = (long)blockIdx.x * blockDim.x + threadIdx.x;
  if (i < n32) {
    float4 e = ((const float4*)emb)[i & 31];
    float4 f = ((const float4*)feat)[i];
    float4 y; y.x = e.x + f.x; y.y = e.y + f.y; y.z = e.z + f.z; y.w = e.w + f.w;
    ((float4*)vx)[i] = y;
  }
}

__global__ void k_add(const float* __restrict__ a, const float* __restrict__ b,
                      float* __restrict__ o, long n32) {
  long i = (long)blockIdx.x * blockDim.x + threadIdx.x;
  if (i < n32) {
    float4 x = ((const float4*)a)[i], y = ((const float4*)b)[i];
    float4 z; z.x = x.x + y.x; z.y = x.y + y.y; z.z = x.z + y.z; z.w = x.w + y.w;
    ((float4*)o)[i] = z;
  }
}

__global__ void k_addrelu(float* __restrict__ vx, const float* __restrict__ z, long n32) {
  long i = (long)blockIdx.x * blockDim.x + threadIdx.x;
  if (i < n32) {
    float4 x = ((float4*)vx)[i], y = ((const float4*)z)[i];
    x.x += fmaxf(y.x, 0.f); x.y += fmaxf(y.y, 0.f);
    x.z += fmaxf(y.z, 0.f); x.w += fmaxf(y.w, 0.f);
    ((float4*)vx)[i] = x;
  }
}

// o[v][:] = st[v][:] + m[batch[v]][:]
__global__ void k_sync(const float* __restrict__ st, const int* __restrict__ bat,
                       const float* __restrict__ m, float* __restrict__ o, long n32) {
  long i = (long)blockIdx.x * blockDim.x + threadIdx.x;
  if (i < n32) {
    long v = i >> 5; int c4 = (int)(i & 31);
    float4 a = ((const float4*)st)[i];
    float4 b = ((const float4*)(m + (long)bat[v] * HID))[c4];
    float4 y; y.x = a.x + b.x; y.y = a.y + b.y; y.z = a.z + b.z; y.w = a.w + b.w;
    ((float4*)o)[i] = y;
  }
}

// ---------------------------------------------------------------- host
static inline int nblk(long n) { return (int)((n + 255) / 256); }

extern "C" void kernel_launch(void* const* d_in, const int* in_sizes, int n_in,
                              void* d_out, int out_size, void* d_ws, size_t ws_size,
                              hipStream_t stream) {
  const float* drug_x    = (const float*)d_in[0];
  const float* prot_x    = (const float*)d_in[1];
  const float* d_feature = (const float*)d_in[2];
  const float* p_feature = (const float*)d_in[3];
  const int*   d_ei      = (const int*)d_in[4];
  const int*   d_batch   = (const int*)d_in[5];
  const int*   d_sn      = (const int*)d_in[6];
  const int*   d_si      = (const int*)d_in[7];
  const int*   p_ei      = (const int*)d_in[8];
  const int*   p_batch   = (const int*)d_in[9];
  const int*   p_sn      = (const int*)d_in[10];
  const int*   p_si      = (const int*)d_in[11];
  const float* enc_Wd    = (const float*)d_in[12];
  const float* enc_bd    = (const float*)d_in[13];
  const float* enc_Wp    = (const float*)d_in[14];
  const float* enc_bp    = (const float*)d_in[15];
  const float* convW_d   = (const float*)d_in[16];
  const float* convb_d   = (const float*)d_in[17];
  const float* convW_p   = (const float*)d_in[18];
  const float* convb_p   = (const float*)d_in[19];
  const float* bng_d     = (const float*)d_in[20];
  const float* bnb_d     = (const float*)d_in[21];
  const float* bng_p     = (const float*)d_in[22];
  const float* bnb_p     = (const float*)d_in[23];
  const float* vn_emb_d  = (const float*)d_in[24];
  const float* vn_emb_p  = (const float*)d_in[25];
  const float* vnW_d     = (const float*)d_in[26];
  const float* vnb_d     = (const float*)d_in[27];
  const float* vnW_p     = (const float*)d_in[28];
  const float* vnb_p     = (const float*)d_in[29];
  const float* mlpd1_W   = (const float*)d_in[30];
  const float* mlpd1_b   = (const float*)d_in[31];
  const float* mlpd1_g   = (const float*)d_in[32];
  const float* mlpd1_be  = (const float*)d_in[33];
  const float* mlpp1_W   = (const float*)d_in[34];
  const float* mlpp1_b   = (const float*)d_in[35];
  const float* mlpp1_g   = (const float*)d_in[36];
  const float* mlpp1_be  = (const float*)d_in[37];

  const int ND = in_sizes[5];
  const long ED = in_sizes[4] / 2;
  const long SD = in_sizes[6];
  const int NP = in_sizes[9];
  const long EP = in_sizes[8] / 2;
  const long SP = in_sizes[10];
  const int B  = in_sizes[2] / HID;
  const int Kd = in_sizes[0] / ND;
  const int Kp = in_sizes[1] / NP;
  const long b32 = (long)B * 32;

  // ---- workspace layout (floats)
  float* w = (float*)d_ws;
  size_t off = 0;
  auto walloc = [&](size_t n) { float* p = w + off; off += n; return p; };
  float* hp     = walloc((size_t)NP * HID);
  float* hwp    = walloc((size_t)NP * HID);
  float* stp    = walloc((size_t)NP * HID);
  float* hd     = walloc((size_t)ND * HID);
  float* hwd    = walloc((size_t)ND * HID);
  float* stdd   = walloc((size_t)ND * HID);
  float* nrm_p  = walloc(NP);
  float* nrm_d  = walloc(ND);
  float* cnt_sp = walloc(NP);
  float* cnt_sd = walloc(ND);
  float* cntb_d = walloc(B);
  float* cntb_p = walloc(B);
  float* coef_d = walloc(ED);
  float* coef_p = walloc(EP);
  float* vx_d   = walloc((size_t)B * HID);
  float* vx_p   = walloc((size_t)B * HID);
  float* bufA   = walloc((size_t)B * HID);
  float* bufB   = walloc((size_t)B * HID);
  float* bufC   = walloc((size_t)B * HID);
  float* bufD   = walloc((size_t)B * HID);
  float* stats  = walloc(2 * HID);

  auto zero = [&](float* p, long n) {
    int bl = nblk(n); if (bl > 32768) bl = 32768;
    k_zero<<<bl, 256, 0, stream>>>(p, n);
  };

  // ---- structure-dependent constants (deterministic per call)
  zero(nrm_d, ND); zero(nrm_p, NP);
  k_deg<<<nblk(ED), 256, 0, stream>>>(d_ei + ED, ED, nrm_d);
  k_deg<<<nblk(EP), 256, 0, stream>>>(p_ei + EP, EP, nrm_p);
  k_nrm<<<nblk(ND), 256, 0, stream>>>(nrm_d, ND);
  k_nrm<<<nblk(NP), 256, 0, stream>>>(nrm_p, NP);
  k_ecoef<<<nblk(ED), 256, 0, stream>>>(d_ei, d_ei + ED, ED, nrm_d, coef_d);
  k_ecoef<<<nblk(EP), 256, 0, stream>>>(p_ei, p_ei + EP, EP, nrm_p, coef_p);
  zero(cnt_sd, ND); zero(cnt_sp, NP);
  k_countf<<<nblk(SD), 256, 0, stream>>>(d_si, SD, cnt_sd);
  k_countf<<<nblk(SP), 256, 0, stream>>>(p_si, SP, cnt_sp);
  zero(cntb_d, B); zero(cntb_p, B);
  k_countf<<<nblk(ND), 256, 0, stream>>>(d_batch, ND, cntb_d);
  k_countf<<<nblk(NP), 256, 0, stream>>>(p_batch, NP, cntb_p);

  // ---- encoders + virtual-node init
  k_encode<<<ND, 128, 0, stream>>>(drug_x, Kd, enc_Wd, enc_bd, hd);
  k_encode<<<NP, 128, 0, stream>>>(prot_x, Kp, enc_Wp, enc_bp, hp);
  k_vxinit<<<nblk(b32), 256, 0, stream>>>(vn_emb_d, d_feature, vx_d, b32);
  k_vxinit<<<nblk(b32), 256, 0, stream>>>(vn_emb_p, p_feature, vx_p, b32);

  // ---- per-layer, per-graph body
  auto run_side = [&](float* h, float* hw, float* st, int N, const int* ei, long E,
                      const float* coef, const int* batch,
                      const int* sn, const int* si, long S,
                      const float* convW, const float* convb,
                      const float* bng, const float* bnb,
                      const float* nrm, const float* cnt_s,
                      float* vx, const float* vnW, const float* vnb, int layer) {
    const int* src = ei; const int* dst = ei + E;
    const long n32 = (long)N * 32, n128 = (long)N * HID;
    // hw = (h + vx[batch]) @ W[layer]           (WMMA)
    k_gemm128<<<N / 64, 256, 0, stream>>>(h, convW + (size_t)layer * HID * HID,
                                          nullptr, vx, batch, hw, 0);
    // h = hw * nrm^2 + b  ;  h[dst] += hw[src] * coef[e]
    k_self<<<nblk(n32), 256, 0, stream>>>(hw, nrm, convb + layer * HID, h, n32);
    k_edges<<<nblk(E * 32), 256, 0, stream>>>(src, dst, coef, E, hw, h);
    // h = relu(bn(h))
    zero(stats, 2 * HID);
    k_bnstats<<<(N + 63) / 64, 128, 0, stream>>>(h, N, stats);
    k_bnapply<<<nblk(n32), 256, 0, stream>>>(h, n32, 1.0f / (float)N, stats,
                                             bng + layer * HID, bnb + layer * HID, 1);
    // st = seg_mean(h[sn], si, N)
    zero(st, n128);
    k_gsadd<<<nblk(S * 32), 256, 0, stream>>>(sn, si, S, h, st);
    k_divcnt<<<nblk(n32), 256, 0, stream>>>(st, cnt_s, n32);
    // virtual-node update (layers 0,1)
    if (layer < 2) {
      zero(bufA, (long)B * HID);
      k_segsum<<<nblk(n32), 256, 0, stream>>>(batch, n32, st, bufA);
      k_add<<<nblk(b32), 256, 0, stream>>>(bufA, vx, bufB, b32);
      k_gemm128<<<B / 64, 256, 0, stream>>>(bufB, vnW + (size_t)layer * HID * HID,
                                            vnb + layer * HID, nullptr, nullptr, bufC, 0);
      k_addrelu<<<nblk(b32), 256, 0, stream>>>(vx, bufC, b32);
    }
  };

  for (int i = 0; i < 3; ++i) {
    run_side(hd, hwd, stdd, ND, d_ei, ED, coef_d, d_batch, d_sn, d_si, SD,
             convW_d, convb_d, bng_d, bnb_d, nrm_d, cnt_sd, vx_d, vnW_d, vnb_d, i);
    run_side(hp, hwp, stp, NP, p_ei, EP, coef_p, p_batch, p_sn, p_si, SP,
             convW_p, convb_p, bng_p, bnb_p, nrm_p, cnt_sp, vx_p, vnW_p, vnb_p, i);
  }

  const long nd32 = (long)ND * 32, np32 = (long)NP * 32;
  // mp = seg_mean(stp, p_batch) -> bufA ; sync_prot (drug nodes) -> hwd
  zero(bufA, (long)B * HID);
  k_segsum<<<nblk(np32), 256, 0, stream>>>(p_batch, np32, stp, bufA);
  k_divcnt<<<nblk(b32), 256, 0, stream>>>(bufA, cntb_p, b32);
  k_sync<<<nblk(nd32), 256, 0, stream>>>(stdd, d_batch, bufA, hwd, nd32);
  // md = seg_mean(stdd, d_batch) -> bufB ; sync_drug (prot nodes) -> hwp
  zero(bufB, (long)B * HID);
  k_segsum<<<nblk(nd32), 256, 0, stream>>>(d_batch, nd32, stdd, bufB);
  k_divcnt<<<nblk(b32), 256, 0, stream>>>(bufB, cntb_d, b32);
  k_sync<<<nblk(np32), 256, 0, stream>>>(stp, p_batch, bufB, hwp, np32);
  // u_d = seg_mean(sync_prot, d_batch) -> bufC ; u_p = seg_mean(sync_drug, p_batch) -> bufD
  zero(bufC, (long)B * HID);
  k_segsum<<<nblk(nd32), 256, 0, stream>>>(d_batch, nd32, hwd, bufC);
  k_divcnt<<<nblk(b32), 256, 0, stream>>>(bufC, cntb_d, b32);
  zero(bufD, (long)B * HID);
  k_segsum<<<nblk(np32), 256, 0, stream>>>(p_batch, np32, hwp, bufD);
  k_divcnt<<<nblk(b32), 256, 0, stream>>>(bufD, cntb_p, b32);

  // final MLPs: concat([u, feature]) @ W(256x128) == u @ W[:128] + feature @ W[128:]
  float* out_d = (float*)d_out;
  float* out_p = out_d + (long)B * HID;
  k_gemm128<<<B / 64, 256, 0, stream>>>(bufC, mlpd1_W, mlpd1_b, nullptr, nullptr, out_d, 0);
  k_gemm128<<<B / 64, 256, 0, stream>>>(d_feature, mlpd1_W + HID * HID, nullptr, nullptr, nullptr, out_d, 1);
  zero(stats, 2 * HID);
  k_bnstats<<<(B + 63) / 64, 128, 0, stream>>>(out_d, B, stats);
  k_bnapply<<<nblk(b32), 256, 0, stream>>>(out_d, b32, 1.0f / (float)B, stats, mlpd1_g, mlpd1_be, 0);

  k_gemm128<<<B / 64, 256, 0, stream>>>(bufD, mlpp1_W, mlpp1_b, nullptr, nullptr, out_p, 0);
  k_gemm128<<<B / 64, 256, 0, stream>>>(p_feature, mlpp1_W + HID * HID, nullptr, nullptr, nullptr, out_p, 1);
  zero(stats, 2 * HID);
  k_bnstats<<<(B + 63) / 64, 128, 0, stream>>>(out_p, B, stats);
  k_bnapply<<<nblk(b32), 256, 0, stream>>>(out_p, b32, 1.0f / (float)B, stats, mlpp1_g, mlpp1_be, 0);

  (void)n_in; (void)out_size; (void)ws_size;
}
